// BERT_83932250898843
// MI455X (gfx1250) — matrix operand order repaired and checked by reading, not compiled, not run on hardware
//
#include <hip/hip_runtime.h>

// ---------------------------------------------------------------------------
// BERT-base forward for MI455X (gfx1250, wave32, WMMA bf16, async-LDS copies)
// Mixed precision: weights/activations feeding WMMA are bf16; residual/LN/
// pooler path stays fp32.
// ---------------------------------------------------------------------------

typedef __attribute__((ext_vector_type(16))) __bf16 v16bf;
typedef __attribute__((ext_vector_type(8)))  float  v8f;

#define BD   768        // D
#define BS   512        // S
#define BB   16         // B
#define BH   12         // H
#define BL   12         // L
#define BF   3072       // F
#define MT   (BB*BS)    // 8192 token rows
#define HD   64
#define SCALE 0.125f    // 64^-0.5
#define LNEPS 1e-12f

__device__ __forceinline__ v8f wmma_bf16(v16bf a, v16bf b, v8f c) {
    return __builtin_amdgcn_wmma_f32_16x16x32_bf16(false, a, false, b,
                                                   (short)0, c, false, false);
}

// LDS byte offset of a generic pointer known to point into LDS.
__device__ __forceinline__ unsigned ldsOff(const void* p) {
    return (unsigned)(unsigned long long)
           (__attribute__((address_space(3))) const char*)p;
}

// Async 16B global -> LDS copy (per-lane), tracked by ASYNCcnt.
__device__ __forceinline__ void async_b128(unsigned lds, const void* gptr) {
    asm volatile("global_load_async_to_lds_b128 %0, %1, off"
                 :: "v"(lds), "v"(gptr) : "memory");
}

// ---------------------------------------------------------------------------
// fp32 -> bf16 bulk conversion (weights; once per launch).  n % 8 == 0.
// ---------------------------------------------------------------------------
__global__ __launch_bounds__(256)
void cvt_bf16_kernel(const float* __restrict__ in, __bf16* __restrict__ out,
                     long n)
{
    long i = ((long)blockIdx.x * 256 + threadIdx.x) * 8;
    if (i >= n) return;
    const float4* p = reinterpret_cast<const float4*>(in + i);
    float4 f0 = p[0], f1 = p[1];
    __bf16* o = out + i;
    o[0] = (__bf16)f0.x; o[1] = (__bf16)f0.y;
    o[2] = (__bf16)f0.z; o[3] = (__bf16)f0.w;
    o[4] = (__bf16)f1.x; o[5] = (__bf16)f1.y;
    o[6] = (__bf16)f1.z; o[7] = (__bf16)f1.w;
}

// ---------------------------------------------------------------------------
// Embedding gather + LayerNorm -> fp32 master x and bf16 shadow.
// ---------------------------------------------------------------------------
__global__ __launch_bounds__(256)
void embed_ln_kernel(const int* __restrict__ tokens,
                     const float* __restrict__ tok_emb,
                     const float* __restrict__ pos_emb,
                     const float* __restrict__ sent_emb,
                     const float* __restrict__ w,
                     const float* __restrict__ bvec,
                     float* __restrict__ xf, __bf16* __restrict__ xb)
{
    const int row = blockIdx.x;
    const int s   = row % BS;
    const int tok = tokens[row];
    const int tid = threadIdx.x;
    __shared__ float red[256];

    float v[3];
    #pragma unroll
    for (int i = 0; i < 3; ++i) {
        int d = tid + i * 256;
        v[i] = tok_emb[(size_t)tok * BD + d] + pos_emb[(size_t)s * BD + d] + sent_emb[d];
    }
    float sum = v[0] + v[1] + v[2];
    red[tid] = sum; __syncthreads();
    for (int st = 128; st > 0; st >>= 1) {
        if (tid < st) red[tid] += red[tid + st];
        __syncthreads();
    }
    const float mu = red[0] * (1.0f / BD);
    __syncthreads();
    float vs = 0.f;
    #pragma unroll
    for (int i = 0; i < 3; ++i) { float d0 = v[i] - mu; vs += d0 * d0; }
    red[tid] = vs; __syncthreads();
    for (int st = 128; st > 0; st >>= 1) {
        if (tid < st) red[tid] += red[tid + st];
        __syncthreads();
    }
    const float rs = rsqrtf(red[0] * (1.0f / BD) + LNEPS);
    #pragma unroll
    for (int i = 0; i < 3; ++i) {
        int d = tid + i * 256;
        float o = (v[i] - mu) * rs * w[d] + bvec[d];
        xf[(size_t)row * BD + d] = o;
        xb[(size_t)row * BD + d] = (__bf16)o;
    }
}

// ---------------------------------------------------------------------------
// x = LN(x + t);  writes fp32 master and bf16 shadow.
// ---------------------------------------------------------------------------
__global__ __launch_bounds__(256)
void add_ln_kernel(float* __restrict__ xf, __bf16* __restrict__ xb,
                   const float* __restrict__ t,
                   const float* __restrict__ w,
                   const float* __restrict__ bvec)
{
    const int row = blockIdx.x;
    const int tid = threadIdx.x;
    __shared__ float red[256];

    float v[3];
    #pragma unroll
    for (int i = 0; i < 3; ++i) {
        int d = tid + i * 256;
        v[i] = xf[(size_t)row * BD + d] + t[(size_t)row * BD + d];
    }
    float sum = v[0] + v[1] + v[2];
    red[tid] = sum; __syncthreads();
    for (int st = 128; st > 0; st >>= 1) {
        if (tid < st) red[tid] += red[tid + st];
        __syncthreads();
    }
    const float mu = red[0] * (1.0f / BD);
    __syncthreads();
    float vs = 0.f;
    #pragma unroll
    for (int i = 0; i < 3; ++i) { float d0 = v[i] - mu; vs += d0 * d0; }
    red[tid] = vs; __syncthreads();
    for (int st = 128; st > 0; st >>= 1) {
        if (tid < st) red[tid] += red[tid + st];
        __syncthreads();
    }
    const float rs = rsqrtf(red[0] * (1.0f / BD) + LNEPS);
    #pragma unroll
    for (int i = 0; i < 3; ++i) {
        int d = tid + i * 256;
        float o = (v[i] - mu) * rs * w[d] + bvec[d];
        xf[(size_t)row * BD + d] = o;
        xb[(size_t)row * BD + d] = (__bf16)o;
    }
}

// ---------------------------------------------------------------------------
// GEMM: C[M,N] = A[M,K] @ W[N,K]^T + bias.  A,W bf16; out fp32 or bf16.
// Block 256 thr (8 waves), tile 128x128; wave tile 32x64 (2x4 WMMA frags).
// Double-buffered LDS fed by global_load_async_to_lds_b128 (ASYNCcnt).
// Requires M%128==0, N%128==0, K%32==0.
// flags: bit0 = exact GELU, bit1 = bf16 output (else fp32).
// ---------------------------------------------------------------------------
__global__ __launch_bounds__(256)
void gemm_bf16_kernel(const __bf16* __restrict__ A,
                      const __bf16* __restrict__ W,
                      const float* __restrict__ bias,
                      float* __restrict__ Cf,
                      __bf16* __restrict__ Cb,
                      int M, int N, int K, int flags)
{
    __shared__ __align__(16) __bf16 As[2][128][48];
    __shared__ __align__(16) __bf16 Ws[2][128][48];

    const int tid  = threadIdx.x;
    const int lane = tid & 31;
    const int wid  = tid >> 5;
    const int wm   = wid & 3;          // 0..3  -> 32-row band
    const int wn   = wid >> 2;         // 0..1  -> 64-col band
    const int lo   = lane & 15;
    const int hi   = lane >> 4;
    const int blockN = blockIdx.x * 128;
    const int blockM = blockIdx.y * 128;

    // per-thread async copy slots: rows (tid>>2, tid>>2+64), 8 bf16 at col (tid&3)*8
    const int cr = tid >> 2;           // 0..63
    const int cc = (tid & 3) * 8;      // 0/8/16/24
    const int kTiles = K >> 5;

    auto issue = [&](int t, int buf) {
        const __bf16* ga = A + (size_t)(blockM + cr) * K + t * 32 + cc;
        const __bf16* gw = W + (size_t)(blockN + cr) * K + t * 32 + cc;
        async_b128(ldsOff(&As[buf][cr][cc]),      ga);
        async_b128(ldsOff(&As[buf][cr + 64][cc]), ga + (size_t)64 * K);
        async_b128(ldsOff(&Ws[buf][cr][cc]),      gw);
        async_b128(ldsOff(&Ws[buf][cr + 64][cc]), gw + (size_t)64 * K);
    };

    v8f c[2][4] = {};
    issue(0, 0);

    for (int t = 0; t < kTiles; ++t) {
        const int p = t & 1;
        if (t + 1 < kTiles) {
            issue(t + 1, p ^ 1);
            asm volatile("s_wait_asynccnt 0x4" ::: "memory");  // tile t resident
        } else {
            asm volatile("s_wait_asynccnt 0x0" ::: "memory");
        }
        __syncthreads();

        v16bf a[2], b[4];
        #pragma unroll
        for (int mt = 0; mt < 2; ++mt) {
            const int row = wm * 32 + mt * 16 + lo;
            #pragma unroll
            for (int e = 0; e < 16; ++e) {
                const int kl = ((e & 8) << 1) | (hi << 3) | (e & 7);
                a[mt][e] = As[p][row][kl];
            }
        }
        #pragma unroll
        for (int nt = 0; nt < 4; ++nt) {
            const int col = wn * 64 + nt * 16 + lo;
            #pragma unroll
            for (int e = 0; e < 16; ++e)
                b[nt][e] = Ws[p][col][((e & 8) << 1) | (hi << 3) | (e & 7)];
        }
        #pragma unroll
        for (int mt = 0; mt < 2; ++mt)
            #pragma unroll
            for (int nt = 0; nt < 4; ++nt)
                c[mt][nt] = wmma_bf16(a[mt], b[nt], c[mt][nt]);

        __syncthreads();   // all reads of buffer p done before it is refilled
    }

    const int doGelu = flags & 1;
    const int bfOut  = flags & 2;
    #pragma unroll
    for (int mt = 0; mt < 2; ++mt) {
        #pragma unroll
        for (int nt = 0; nt < 4; ++nt) {
            const int gr = blockM + wm * 32 + mt * 16 + (hi << 3);
            const int gc = blockN + wn * 64 + nt * 16 + lo;
            const float bv = bias[gc];
            #pragma unroll
            for (int r = 0; r < 8; ++r) {
                float val = c[mt][nt][r] + bv;
                if (doGelu)
                    val = 0.5f * val * (1.0f + erff(val * 0.70710678118654752f));
                if (bfOut)
                    Cb[(size_t)(gr + r) * N + gc] = (__bf16)val;
                else
                    Cf[(size_t)(gr + r) * N + gc] = val;
            }
        }
    }
}

// ---------------------------------------------------------------------------
// WMMA flash attention.  Grid (S/64, H, B), block 128 thr (4 waves).
// bf16 qkv in, bf16 attention-out.  Online softmax.  KV tiles (32 keys)
// double-buffered in LDS via global_load_async_to_lds_b128 so the next
// tile's fetch overlaps this tile's WMMA + softmax work.
// ---------------------------------------------------------------------------
__global__ __launch_bounds__(128)
void attention_kernel(const __bf16* __restrict__ qkv, __bf16* __restrict__ out)
{
    __shared__ __align__(16) __bf16 Ks[2][32][72];
    __shared__ __align__(16) __bf16 Vs[2][32][72];
    __shared__ __align__(16) __bf16 Pb[4][16][48];

    const int qbase = blockIdx.x * 64;
    const int h     = blockIdx.y;
    const int b     = blockIdx.z;
    const int tid   = threadIdx.x;
    const int lane  = tid & 31;
    const int wid   = tid >> 5;
    const int lo    = lane & 15;
    const int hi    = lane >> 4;

    const size_t base = (size_t)b * BS * (3 * BD);

    // Q fragments (16 rows x 64 ch -> two 16x32 bf16 A-frags)
    v16bf aq[2];
    {
        const int m = qbase + wid * 16 + lo;
        const __bf16* qrow = qkv + base + (size_t)m * (3 * BD) + h * HD;
        #pragma unroll
        for (int kk = 0; kk < 2; ++kk)
            #pragma unroll
            for (int e = 0; e < 16; ++e) {
                const int ch = kk * 32 + (((e & 8) << 1) | (hi << 3) | (e & 7));
                aq[kk][e] = qrow[ch];
            }
    }

    float rowmax[8], rowsum[8];
    #pragma unroll
    for (int r = 0; r < 8; ++r) { rowmax[r] = -1e30f; rowsum[r] = 0.f; }
    v8f o[4] = {};

    const int krow = tid >> 2;          // 0..31 key row within tile
    const int kseg = (tid & 3) * 16;    // 16 bf16 = 32B per thread

    auto issueKV = [&](int t, int buf) {
        const int key = t * 32 + krow;
        const __bf16* kp = qkv + base + (size_t)key * (3 * BD) + BD + h * HD + kseg;
        async_b128(ldsOff(&Ks[buf][krow][kseg]),     kp);
        async_b128(ldsOff(&Ks[buf][krow][kseg + 8]), kp + 8);
        async_b128(ldsOff(&Vs[buf][krow][kseg]),     kp + BD);
        async_b128(ldsOff(&Vs[buf][krow][kseg + 8]), kp + BD + 8);
    };

    issueKV(0, 0);

    for (int t = 0; t < BS / 32; ++t) {
        const int p = t & 1;
        if (t + 1 < BS / 32) {
            issueKV(t + 1, p ^ 1);
            asm volatile("s_wait_asynccnt 0x4" ::: "memory");  // tile t resident
        } else {
            asm volatile("s_wait_asynccnt 0x0" ::: "memory");
        }
        __syncthreads();

        // logits 16x32 = Q @ Ktile^T
        v8f s[2] = {};
        #pragma unroll
        for (int ns = 0; ns < 2; ++ns) {
            #pragma unroll
            for (int kk = 0; kk < 2; ++kk) {
                v16bf bk;
                #pragma unroll
                for (int e = 0; e < 16; ++e)
                    bk[e] = Ks[p][ns * 16 + lo][kk * 32 + (((e & 8) << 1) | (hi << 3) | (e & 7))];
                s[ns] = wmma_bf16(aq[kk], bk, s[ns]);
            }
        }

        // online softmax over the 16-lane halves
        float tmax[8];
        #pragma unroll
        for (int r = 0; r < 8; ++r) {
            s[0][r] *= SCALE; s[1][r] *= SCALE;
            tmax[r] = fmaxf(s[0][r], s[1][r]);
        }
        #pragma unroll
        for (int msk = 1; msk < 16; msk <<= 1)
            #pragma unroll
            for (int r = 0; r < 8; ++r)
                tmax[r] = fmaxf(tmax[r], __shfl_xor(tmax[r], msk, 32));

        float psum[8];
        #pragma unroll
        for (int r = 0; r < 8; ++r) {
            const float nm = fmaxf(rowmax[r], tmax[r]);
            const float al = __expf(rowmax[r] - nm);
            rowmax[r] = nm;
            const float e0 = __expf(s[0][r] - nm);
            const float e1 = __expf(s[1][r] - nm);
            s[0][r] = e0; s[1][r] = e1;
            psum[r] = e0 + e1;
            rowsum[r] *= al;
            #pragma unroll
            for (int nv = 0; nv < 4; ++nv) o[nv][r] *= al;
        }
        #pragma unroll
        for (int msk = 1; msk < 16; msk <<= 1)
            #pragma unroll
            for (int r = 0; r < 8; ++r)
                psum[r] += __shfl_xor(psum[r], msk, 32);
        #pragma unroll
        for (int r = 0; r < 8; ++r) rowsum[r] += psum[r];

        // P: C layout -> per-wave LDS -> A layout
        #pragma unroll
        for (int ns = 0; ns < 2; ++ns)
            #pragma unroll
            for (int r = 0; r < 8; ++r)
                Pb[wid][r + (hi << 3)][ns * 16 + lo] = (__bf16)s[ns][r];

        v16bf pa;
        #pragma unroll
        for (int e = 0; e < 16; ++e) {
            const int kl = ((e & 8) << 1) | (hi << 3) | (e & 7);
            pa[e] = Pb[wid][lo][kl];
        }

        // O += P @ Vtile
        #pragma unroll
        for (int nv = 0; nv < 4; ++nv) {
            v16bf bv;
            #pragma unroll
            for (int e = 0; e < 16; ++e)
                bv[e] = Vs[p][((e & 8) << 1) | (hi << 3) | (e & 7)][nv * 16 + lo];
            o[nv] = wmma_bf16(pa, bv, o[nv]);
        }

        __syncthreads();   // all reads of buffer p done before it is refilled
    }

    #pragma unroll
    for (int nv = 0; nv < 4; ++nv) {
        #pragma unroll
        for (int r = 0; r < 8; ++r) {
            const int mr = qbase + wid * 16 + (hi << 3) + r;
            out[((size_t)(b * BS + mr)) * BD + h * HD + nv * 16 + lo] =
                (__bf16)(o[nv][r] / rowsum[r]);
        }
    }
}

// ---------------------------------------------------------------------------
// Pooler: out[b,j] = tanh(x[b,0,:] . pool_w[j,:] + pool_b[j])  (fp32)
// ---------------------------------------------------------------------------
__global__ __launch_bounds__(256)
void pooler_kernel(const float* __restrict__ x,
                   const float* __restrict__ pw,
                   const float* __restrict__ pb,
                   float* __restrict__ out)
{
    const int g = blockIdx.x * 256 + threadIdx.x;
    const int b = g / BD;
    const int j = g % BD;
    const float* xr = x + (size_t)b * BS * BD;
    const float* wr = pw + (size_t)j * BD;
    float acc = pb[j];
    for (int k = 0; k < BD; ++k) acc += xr[k] * wr[k];
    out[g] = tanhf(acc);
}

// ---------------------------------------------------------------------------
// Host-side orchestration
// ---------------------------------------------------------------------------
extern "C" void kernel_launch(void* const* d_in, const int* in_sizes, int n_in,
                              void* d_out, int out_size, void* d_ws, size_t ws_size,
                              hipStream_t stream)
{
    const int*   tokens   = (const int*)  d_in[0];
    const float* tok_emb  = (const float*)d_in[1];
    const float* pos_emb  = (const float*)d_in[2];
    const float* sent_emb = (const float*)d_in[3];
    const float* emb_ln_w = (const float*)d_in[4];
    const float* emb_ln_b = (const float*)d_in[5];
    const float* qkv_w    = (const float*)d_in[6];
    const float* qkv_b    = (const float*)d_in[7];
    const float* proj_w   = (const float*)d_in[8];
    const float* proj_b   = (const float*)d_in[9];
    const float* ln1_w    = (const float*)d_in[10];
    const float* ln1_b    = (const float*)d_in[11];
    const float* ln2_w    = (const float*)d_in[12];
    const float* ln2_b    = (const float*)d_in[13];
    const float* mlp_w1   = (const float*)d_in[14];
    const float* mlp_b1   = (const float*)d_in[15];
    const float* mlp_w2   = (const float*)d_in[16];
    const float* mlp_b2   = (const float*)d_in[17];
    const float* pool_w   = (const float*)d_in[18];
    const float* pool_b   = (const float*)d_in[19];

    char* wp = (char*)d_ws;
    auto carve = [&](size_t bytes) {
        char* p = wp;
        wp += (bytes + 255) & ~(size_t)255;
        return p;
    };
    float*  xf    = (float*) carve((size_t)MT * BD * 4);
    float*  tmp   = (float*) carve((size_t)MT * BD * 4);
    __bf16* xb    = (__bf16*)carve((size_t)MT * BD * 2);
    __bf16* qkvb  = (__bf16*)carve((size_t)MT * 3 * BD * 2);
    __bf16* attnb = (__bf16*)carve((size_t)MT * BD * 2);
    __bf16* hb    = (__bf16*)carve((size_t)MT * BF * 2);
    __bf16* wqkv  = (__bf16*)carve((size_t)BL * 3 * BD * BD * 2);
    __bf16* wproj = (__bf16*)carve((size_t)BL * BD * BD * 2);
    __bf16* w1b   = (__bf16*)carve((size_t)BL * BF * BD * 2);
    __bf16* w2b   = (__bf16*)carve((size_t)BL * BD * BF * 2);

    // weights -> bf16 (once per launch)
    {
        long n;
        n = (long)BL * 3 * BD * BD;
        cvt_bf16_kernel<<<(unsigned)(n / 8 / 256), 256, 0, stream>>>(qkv_w, wqkv, n);
        n = (long)BL * BD * BD;
        cvt_bf16_kernel<<<(unsigned)(n / 8 / 256), 256, 0, stream>>>(proj_w, wproj, n);
        n = (long)BL * BF * BD;
        cvt_bf16_kernel<<<(unsigned)(n / 8 / 256), 256, 0, stream>>>(mlp_w1, w1b, n);
        n = (long)BL * BD * BF;
        cvt_bf16_kernel<<<(unsigned)(n / 8 / 256), 256, 0, stream>>>(mlp_w2, w2b, n);
    }

    embed_ln_kernel<<<MT, 256, 0, stream>>>(tokens, tok_emb, pos_emb, sent_emb,
                                            emb_ln_w, emb_ln_b, xf, xb);

    for (int l = 0; l < BL; ++l) {
        const __bf16* qw = wqkv  + (size_t)l * 3 * BD * BD;
        const __bf16* pw = wproj + (size_t)l * BD * BD;
        const __bf16* w1 = w1b   + (size_t)l * BF * BD;
        const __bf16* w2 = w2b   + (size_t)l * BD * BF;

        // QKV projection -> bf16
        gemm_bf16_kernel<<<dim3(3 * BD / 128, MT / 128), 256, 0, stream>>>(
            xb, qw, qkv_b + (size_t)l * 3 * BD, nullptr, qkvb,
            MT, 3 * BD, BD, /*flags=*/2);

        attention_kernel<<<dim3(BS / 64, BH, BB), 128, 0, stream>>>(qkvb, attnb);

        // output projection -> fp32 tmp
        gemm_bf16_kernel<<<dim3(BD / 128, MT / 128), 256, 0, stream>>>(
            attnb, pw, proj_b + (size_t)l * BD, tmp, nullptr,
            MT, BD, BD, /*flags=*/0);

        add_ln_kernel<<<MT, 256, 0, stream>>>(xf, xb, tmp,
            ln1_w + (size_t)l * BD, ln1_b + (size_t)l * BD);

        // MLP fc1 + exact GELU -> bf16 hidden
        gemm_bf16_kernel<<<dim3(BF / 128, MT / 128), 256, 0, stream>>>(
            xb, w1, mlp_b1 + (size_t)l * BF, nullptr, hb,
            MT, BF, BD, /*flags=*/3);

        // MLP fc2 -> fp32 tmp
        gemm_bf16_kernel<<<dim3(BD / 128, MT / 128), 256, 0, stream>>>(
            hb, w2, mlp_b2 + (size_t)l * BD, tmp, nullptr,
            MT, BD, BF, /*flags=*/0);

        add_ln_kernel<<<MT, 256, 0, stream>>>(xf, xb, tmp,
            ln2_w + (size_t)l * BD, ln2_b + (size_t)l * BD);
    }

    pooler_kernel<<<(BB * BD) / 256, 256, 0, stream>>>(xf, pool_w, pool_b,
                                                       (float*)d_out);
}
